// DecoderForGeoLossLess_54932631716251
// MI455X (gfx1250) — matrix operand order repaired and checked by reading, not compiled, not run on hardware
//
#include <hip/hip_runtime.h>
#include <hip/hip_bf16.h>

typedef __attribute__((ext_vector_type(16))) __bf16 v16bf;
typedef __attribute__((ext_vector_type(8)))  float  v8f;

// ---------------------------------------------------------------------------
// WMMA helpers (CDNA5 gfx1250, wave32): D = A(16x32 bf16) x B(32x16 bf16) + C
// ---------------------------------------------------------------------------
__device__ __forceinline__ v8f wmma_bf16(v16bf a, v16bf b, v8f c) {
  // (neg_a, A, neg_b, B, c_mod, C, reuse_a, reuse_b)
  return __builtin_amdgcn_wmma_f32_16x16x32_bf16(false, a, false, b, (short)0, c,
                                                 false, false);
}

// A-fragment (16x32 bf16) per ISA layout:
//   lane L holds row M = L&15; elements 0..7  = K (kh*32 + koff .. +7)
//                               elements 8..15 = K (kh*32 + 16 + koff .. +7)
//   koff = (L>>4)*8.  Source row is fp32; convert to bf16 in-register.
__device__ __forceinline__ v16bf load_a(const float* __restrict__ row, int kh, int koff) {
  const float4* p0 = (const float4*)(row + kh * 32 + koff);
  const float4* p1 = (const float4*)(row + kh * 32 + 16 + koff);
  float4 a0 = p0[0], a1 = p0[1];
  float4 b0 = p1[0], b1 = p1[1];
  v16bf r;
  r[0]  = (__bf16)a0.x; r[1]  = (__bf16)a0.y; r[2]  = (__bf16)a0.z; r[3]  = (__bf16)a0.w;
  r[4]  = (__bf16)a1.x; r[5]  = (__bf16)a1.y; r[6]  = (__bf16)a1.z; r[7]  = (__bf16)a1.w;
  r[8]  = (__bf16)b0.x; r[9]  = (__bf16)b0.y; r[10] = (__bf16)b0.z; r[11] = (__bf16)b0.w;
  r[12] = (__bf16)b1.x; r[13] = (__bf16)b1.y; r[14] = (__bf16)b1.z; r[15] = (__bf16)b1.w;
  return r;
}

// B fragments are pre-packed so lane L reads 16 contiguous bf16 (32B).
__device__ __forceinline__ v16bf load_b(const __bf16* __restrict__ Wt, int fragIdx) {
  return *((const v16bf*)(Wt + (long)fragIdx * 16));
}

__device__ __forceinline__ float sigm(float x) { return 1.0f / (1.0f + __expf(-x)); }
__device__ __forceinline__ float tanh_fast(float x) {
  float c = fminf(fmaxf(x, -15.0f), 15.0f);
  float e = __expf(2.0f * c);
  return (e - 1.0f) / (e + 1.0f);
}

// gates accumulation for one 16x16 tile: hx@W_h + x@W_i over K=64 (2 WMMA each)
__device__ __forceinline__ v8f gemm_gate(v16bf Ah0, v16bf Ah1, v16bf Ax0, v16bf Ax1,
                                         const __bf16* __restrict__ Wh,
                                         const __bf16* __restrict__ Wi,
                                         int k, int nt, int L) {
  v8f acc = {};
  int b0 = ((k * 16 + nt) * 2 + 0) * 32 + L;
  int b1 = ((k * 16 + nt) * 2 + 1) * 32 + L;
  acc = wmma_bf16(Ah0, load_b(Wh, b0), acc);
  acc = wmma_bf16(Ah1, load_b(Wh, b1), acc);
  acc = wmma_bf16(Ax0, load_b(Wi, b0), acc);
  acc = wmma_bf16(Ax1, load_b(Wi, b1), acc);
  return acc;
}

__device__ __forceinline__ v8f gemm_cx(v16bf Ac0, v16bf Ac1,
                                       const __bf16* __restrict__ Wc, int k, int q, int L) {
  v8f acc = {};
  acc = wmma_bf16(Ac0, load_b(Wc, ((k * 4 + q) * 2 + 0) * 32 + L), acc);
  acc = wmma_bf16(Ac1, load_b(Wc, ((k * 4 + q) * 2 + 1) * 32 + L), acc);
  return acc;
}

// ---------------------------------------------------------------------------
// Small utility kernels
// ---------------------------------------------------------------------------
__global__ void zerof_kernel(float* p, long n) {
  long i = (long)blockIdx.x * blockDim.x + threadIdx.x;
  if (i < n) p[i] = 0.0f;
}
__global__ void zeroi_kernel(int* p, int n) {
  int i = blockIdx.x * blockDim.x + threadIdx.x;
  if (i < n) p[i] = 0;
}
__global__ void count_kernel(const int* __restrict__ kofs, int n, int* __restrict__ cnt) {
  int i = blockIdx.x * blockDim.x + threadIdx.x;
  if (i < n) atomicAdd(&cnt[kofs[i]], 1);
}
__global__ void scan_kernel(const int* __restrict__ cnt, int* __restrict__ off,
                            int* __restrict__ cur) {
  if (threadIdx.x == 0 && blockIdx.x == 0) {
    int s = 0;
    for (int k = 0; k < 8; ++k) { off[k] = s; cur[k] = s; s += cnt[k]; }
  }
}
__global__ void scatter_kernel(const int* __restrict__ kofs, int n,
                               int* __restrict__ cur, int* __restrict__ bidx) {
  int i = blockIdx.x * blockDim.x + threadIdx.x;
  if (i < n) {
    int pos = atomicAdd(&cur[kofs[i]], 1);
    bidx[pos] = i;
  }
}

// Pre-pack fp32 weights [K][64][Cout] into bf16 B-fragment layout:
//   dest index d = ((((k*NT + nt)*2 + kh)*32 + lane)*16 + e)
//   source      = W[k][ kh*32 + (e>>3)*16 + (lane>>4)*8 + (e&7) ][ nt*16 + (lane&15) ]
__global__ void wtransform_kernel(const float* __restrict__ W, __bf16* __restrict__ Wt,
                                  int Cout) {
  int d = blockIdx.x * blockDim.x + threadIdx.x;
  int NT = Cout >> 4;
  int total = 8 * 64 * Cout;
  if (d >= total) return;
  int e    = d & 15;  int t1 = d >> 4;
  int lane = t1 & 31; int t2 = t1 >> 5;
  int kh   = t2 & 1;  int t3 = t2 >> 1;
  int nt   = t3 % NT; int k  = t3 / NT;
  int row = kh * 32 + (e >> 3) * 16 + (lane >> 4) * 8 + (e & 7);
  int col = nt * 16 + (lane & 15);
  Wt[d] = (__bf16)W[((long)k * 64 + row) * Cout + col];
}

// ---------------------------------------------------------------------------
// Main level kernel: one wave = one 16-row tile inside one kofs-bucket.
// Block = 128 threads = 4 waves = 4 tiles.
// ---------------------------------------------------------------------------
__global__ __launch_bounds__(128)
void lstm_level_kernel(const float* __restrict__ hx_in, const float* __restrict__ cx_in,
                       const float* __restrict__ x_in,
                       const __bf16* __restrict__ Wt_i, const __bf16* __restrict__ Wt_h,
                       const __bf16* __restrict__ Wt_c,
                       const int* __restrict__ parent, const int* __restrict__ bidx,
                       const int* __restrict__ cnt, const int* __restrict__ off,
                       float* __restrict__ hx_out, float* __restrict__ cx_out,
                       int store_cx) {
  const int wave = threadIdx.x >> 5;
  const int L    = threadIdx.x & 31;
  const int tile = blockIdx.x * 4 + wave;

  // Map global tile id -> (bucket k, tile-in-bucket)
  int kbin = -1, tin = 0, tstart = 0, ck = 0, ok = 0;
#pragma unroll
  for (int kk = 0; kk < 8; ++kk) {
    int c = cnt[kk];
    int ntk = (c + 15) >> 4;
    if (kbin < 0 && tile < tstart + ntk) {
      kbin = kk; tin = tile - tstart; ck = c; ok = off[kk];
    }
    tstart += ntk;
  }
  if (kbin < 0) return;

  const int row_base = ok + tin * 16;
  int rows = ck - tin * 16; if (rows > 16) rows = 16;

  // Gather A rows (clamped for padding lanes; stores are predicated).
  const int m    = L & 15;
  const int mc   = (m < rows) ? m : (rows - 1);
  const int j    = bidx[row_base + mc];
  const int p    = parent[j];
  const int koff = (L >> 4) * 8;

  const float* hrow = hx_in + (long)p * 64;
  const float* xrow = x_in  + (long)p * 64;
  const float* crow = cx_in + (long)p * 64;

  v16bf Ah0 = load_a(hrow, 0, koff), Ah1 = load_a(hrow, 1, koff);
  v16bf Ax0 = load_a(xrow, 0, koff), Ax1 = load_a(xrow, 1, koff);
  v16bf Ac0 = load_a(crow, 0, koff), Ac1 = load_a(crow, 1, koff);

  // Output row ids for the 8 C/D rows this lane holds (M = v + (L>=16 ? 8:0)).
  int jr[8];
  const int mbase = (L >> 4) * 8;
#pragma unroll
  for (int v = 0; v < 8; ++v) {
    int mr = mbase + v;
    jr[v] = (mr < rows) ? bidx[row_base + mr] : -1;
  }
  const int ncol = L & 15;

#pragma unroll
  for (int q = 0; q < 4; ++q) {  // hidden-dim 16-col tiles: cols q*16..q*16+15
    v8f gi = gemm_gate(Ah0, Ah1, Ax0, Ax1, Wt_h, Wt_i, kbin,  0 + q, L);
    v8f gf = gemm_gate(Ah0, Ah1, Ax0, Ax1, Wt_h, Wt_i, kbin,  4 + q, L);
    v8f gc = gemm_gate(Ah0, Ah1, Ax0, Ax1, Wt_h, Wt_i, kbin,  8 + q, L);
    v8f go = gemm_gate(Ah0, Ah1, Ax0, Ax1, Wt_h, Wt_i, kbin, 12 + q, L);
    v8f cu = gemm_cx(Ac0, Ac1, Wt_c, kbin, q, L);

#pragma unroll
    for (int v = 0; v < 8; ++v) {
      if (jr[v] >= 0) {
        float ig = sigm(gi[v]);
        float fg = sigm(gf[v]);
        float cg = sigm(gc[v]);  // reference uses sigmoid for cell gate
        float og = sigm(go[v]);
        float cn = fg * cu[v] + ig * cg;
        float hn = og * tanh_fast(cn);
        long  o  = (long)jr[v] * 64 + q * 16 + ncol;
        hx_out[o] = hn;
        if (store_cx) cx_out[o] = cn;
      }
    }
  }
}

// ---------------------------------------------------------------------------
// Host launcher
// ---------------------------------------------------------------------------
extern "C" void kernel_launch(void* const* d_in, const int* in_sizes, int n_in,
                              void* d_out, int out_size, void* d_ws, size_t ws_size,
                              hipStream_t stream) {
  (void)n_in; (void)out_size; (void)ws_size;
  const float* x0 = (const float*)d_in[0];
  const float* x1 = (const float*)d_in[1];
  const float* x2 = (const float*)d_in[2];
  const float* W_i = (const float*)d_in[3];
  const float* W_h = (const float*)d_in[4];
  const float* W_c = (const float*)d_in[5];
  const int* parent0 = (const int*)d_in[6];
  const int* kofs0   = (const int*)d_in[7];
  const int* parent1 = (const int*)d_in[8];
  const int* kofs1   = (const int*)d_in[9];
  const int* parent2 = (const int*)d_in[10];
  const int* kofs2   = (const int*)d_in[11];

  const int N0 = in_sizes[0] / 64;
  const int N1 = in_sizes[6];
  const int N2 = in_sizes[8];
  const int N3 = in_sizes[10];

  // Carve workspace (256B-aligned chunks).
  char* w = (char*)d_ws;
  auto carve = [&](size_t bytes) -> void* {
    void* ptr = (void*)w;
    w += (bytes + 255) & ~(size_t)255;
    return ptr;
  };
  __bf16* Wt_i = (__bf16*)carve((size_t)8 * 64 * 256 * 2);
  __bf16* Wt_h = (__bf16*)carve((size_t)8 * 64 * 256 * 2);
  __bf16* Wt_c = (__bf16*)carve((size_t)8 * 64 * 64 * 2);
  float* Z  = (float*)carve((size_t)N0 * 64 * 4);  // zero hx/cx for level 0
  float* H1 = (float*)carve((size_t)N1 * 64 * 4);
  float* C1 = (float*)carve((size_t)N1 * 64 * 4);
  float* H2 = (float*)carve((size_t)N2 * 64 * 4);
  float* C2 = (float*)carve((size_t)N2 * 64 * 4);
  int maxN = N1 > N2 ? N1 : N2; if (N3 > maxN) maxN = N3;
  int* bidx = (int*)carve((size_t)maxN * 4);
  int* cnt  = (int*)carve(8 * 4);
  int* off  = (int*)carve(8 * 4);
  int* cur  = (int*)carve(8 * 4);

  // Pack weights into bf16 B-fragment layout (once per launch).
  wtransform_kernel<<<(8 * 64 * 256 + 255) / 256, 256, 0, stream>>>(W_i, Wt_i, 256);
  wtransform_kernel<<<(8 * 64 * 256 + 255) / 256, 256, 0, stream>>>(W_h, Wt_h, 256);
  wtransform_kernel<<<(8 * 64 * 64 + 255) / 256, 256, 0, stream>>>(W_c, Wt_c, 64);
  long zn = (long)N0 * 64;
  zerof_kernel<<<(int)((zn + 255) / 256), 256, 0, stream>>>(Z, zn);

  auto level = [&](const float* hxin, const float* cxin, const float* xin,
                   const int* par, const int* ko, int Nout,
                   float* hxout, float* cxout, int store_cx) {
    zeroi_kernel<<<1, 32, 0, stream>>>(cnt, 8);
    count_kernel<<<(Nout + 255) / 256, 256, 0, stream>>>(ko, Nout, cnt);
    scan_kernel<<<1, 1, 0, stream>>>(cnt, off, cur);
    scatter_kernel<<<(Nout + 255) / 256, 256, 0, stream>>>(ko, Nout, cur, bidx);
    int maxtiles = (Nout + 15) / 16 + 8;  // sum of per-bucket ceil() is <= this
    lstm_level_kernel<<<(maxtiles + 3) / 4, 128, 0, stream>>>(
        hxin, cxin, xin, Wt_i, Wt_h, Wt_c, par, bidx, cnt, off, hxout, cxout, store_cx);
  };

  level(Z,  Z,  x0, parent0, kofs0, N1, H1, C1, 1);
  level(H1, C1, x1, parent1, kofs1, N2, H2, C2, 1);
  level(H2, C2, x2, parent2, kofs2, N3, (float*)d_out, nullptr, 0);  // final cx unused
}
